// IFAREModel_recon_5068061409508
// MI455X (gfx1250) — matrix operand order repaired and checked by reading, not compiled, not run on hardware
//
#include <hip/hip_runtime.h>

// ---------------- problem dims ----------------
constexpr int Bn = 2048;
constexpr int Sn = 61;
constexpr int Dn = 768;
constexpr int Hn = 512;
constexpr int Ln = 23;
constexpr int ROWS = Bn * Sn;            // 124928 (== 3904 * 32 exactly)
constexpr int KT = Dn / 32;              // 24 K-tiles of 32 (even -> clean ping-pong)
constexpr int NT = Hn / 16;              // 32 N-tiles of 16
constexpr int MROWS = 32;                // rows per workgroup (2 WMMA row-tiles)
constexpr int LDSV_STRIDE = Dn + 8;      // bf16 elems; 388 dwords % 64 = 4 -> conflict-free

typedef __attribute__((ext_vector_type(16))) __bf16 v16bf;
typedef __attribute__((ext_vector_type(8)))  __bf16 v8bf;
typedef __attribute__((ext_vector_type(8)))  float  v8f;

__device__ __forceinline__ unsigned short f2bf(float f) {
  unsigned int u = __builtin_bit_cast(unsigned int, f);
  u += 0x7FFFu + ((u >> 16) & 1u);       // round-to-nearest-even
  return (unsigned short)(u >> 16);
}

// ---------------------------------------------------------------------------
// Kernel 1: pack fc1_w [512][768] f32 -> bf16 WMMA-B fragments in workspace.
// Fragment (nTile, kTile): 32 lanes x 16 bf16.  B is KxN (K = d-chunk, N = h):
//   lane l (<16):  N = l,     K = kt*32 + e        (e = 0..15)
//   lane l (>=16): N = l-16,  K = kt*32 + 16 + e
// ---------------------------------------------------------------------------
__global__ __launch_bounds__(32) void pack_fc1_kernel(
    const float* __restrict__ fc1_w, unsigned short* __restrict__ fc1pack) {
  const int frag = blockIdx.x;              // 0 .. NT*KT-1
  const int nTile = frag / KT;
  const int kt = frag % KT;
  const int lane = threadIdx.x;
  const int h = nTile * 16 + (lane & 15);
  const int k0 = kt * 32 + ((lane < 16) ? 0 : 16);
  const float* src = fc1_w + (size_t)h * Dn + k0;
  unsigned short* dst = fc1pack + (size_t)frag * 512 + lane * 16;
#pragma unroll
  for (int e = 0; e < 16; ++e) dst[e] = f2bf(src[e]);
}

// ---------------------------------------------------------------------------
// Kernel 2: gate g[row] = fc2_w . tanh(V[row,:] @ fc1_w^T + fc1_b) + fc2_b
// One WG = 32 rows (2 row-tiles), 8 waves x 64 output channels.
// Double-buffered K-loop: 8 v_wmma_f32_16x16x32_bf16 per half-iteration body.
// ---------------------------------------------------------------------------
__global__ __launch_bounds__(256) void gate_kernel(
    const float* __restrict__ V,                 // [ROWS, 768]
    const unsigned short* __restrict__ fc1pack,  // [NT*KT][512] bf16 frags
    const float* __restrict__ fc1_b,             // [512]
    const float* __restrict__ fc2_w,             // [512]
    const float* __restrict__ fc2_b,             // [1]
    float* __restrict__ g_out)                   // [ROWS]
{
  __shared__ __align__(16) unsigned short ldsV[MROWS * LDSV_STRIDE];
  __shared__ float g_lds[MROWS];

  const int tid = threadIdx.x;
  const int lane = tid & 31;
  const int wid = tid >> 5;                      // 0..7
  const size_t rowBase = (size_t)blockIdx.x * MROWS;

  if (tid < MROWS) g_lds[tid] = 0.f;

  // Stage 32x768 f32 -> bf16 into LDS (6144 float4 / 256 threads = 24 each).
  const float4* src4 = (const float4*)(V + rowBase * Dn);
#pragma unroll
  for (int j = 0; j < 24; ++j) {
    const int i = tid + 256 * j;                 // 0..6143
    const int r = i / 192;
    const int c4 = i - r * 192;
    const float4 v = src4[i];
    unsigned short* dst = &ldsV[r * LDSV_STRIDE + c4 * 4];
    dst[0] = f2bf(v.x); dst[1] = f2bf(v.y); dst[2] = f2bf(v.z); dst[3] = f2bf(v.w);
  }
  __syncthreads();

  // A-fragment addressing (contiguous k order, same as packed B):
  //   lane<16 : row = lane,    K = {k0..k0+15}
  //   lane>=16: row = lane-16, K = {k0+16..k0+31}
  const int rowA = lane & 15;
  const int offA = (lane < 16) ? 0 : 16;
  const int nTile0 = wid * 4;                    // 4 N-tiles per wave
  const v16bf* bfrag = (const v16bf*)fc1pack;    // frag = 32 x v16bf

  v8f acc[8] = {};                               // [mt*4 + t], mt = row-tile

  auto loadFrags = [&](int kt, v16bf (&a)[2], v16bf (&b)[4]) {
    const int k0 = kt * 32;
    const unsigned short* ap0 = &ldsV[rowA * LDSV_STRIDE + k0 + offA];
    const unsigned short* ap1 = &ldsV[(rowA + 16) * LDSV_STRIDE + k0 + offA];
    const v8bf l0 = *(const v8bf*)ap0;           // ds_load_b128
    const v8bf h0 = *(const v8bf*)(ap0 + 8);
    const v8bf l1 = *(const v8bf*)ap1;
    const v8bf h1 = *(const v8bf*)(ap1 + 8);
    a[0] = __builtin_shufflevector(l0, h0, 0,1,2,3,4,5,6,7,8,9,10,11,12,13,14,15);
    a[1] = __builtin_shufflevector(l1, h1, 0,1,2,3,4,5,6,7,8,9,10,11,12,13,14,15);
#pragma unroll
    for (int t = 0; t < 4; ++t)
      b[t] = bfrag[((size_t)(nTile0 + t) * KT + kt) * 32 + lane];
  };

  auto doWmma = [&](const v16bf (&a)[2], const v16bf (&b)[4]) {
#pragma unroll
    for (int t = 0; t < 4; ++t) {
      acc[t]     = __builtin_amdgcn_wmma_f32_16x16x32_bf16(
          false, a[0], false, b[t], (short)0, acc[t],     false, false);
      acc[4 + t] = __builtin_amdgcn_wmma_f32_16x16x32_bf16(
          false, a[1], false, b[t], (short)0, acc[4 + t], false, false);
    }
  };

  // Ping-pong double buffer over 24 k-tiles (2 per trip, no register copies).
  v16bf aC[2], bC[4], aN[2], bN[4];
  loadFrags(0, aC, bC);
  for (int kt = 0; kt < KT; kt += 2) {
    loadFrags(kt + 1, aN, bN);                   // issue loads for kt+1
    doWmma(aC, bC);                              // compute kt
    if (kt + 2 < KT) loadFrags(kt + 2, aC, bC);  // issue loads for kt+2
    doWmma(aN, bN);                              // compute kt+1
  }

  // Epilogue: tanh, weight by fc2_w, reduce over 512 columns per row.
  // C/D layout: lane<16 -> N=lane, M=j; lane>=16 -> N=lane-16, M=j+8.
#pragma unroll
  for (int t = 0; t < 4; ++t) {
    const int h = (nTile0 + t) * 16 + (lane & 15);
    const float fb = fc1_b[h];
    const float fw = fc2_w[h];
#pragma unroll
    for (int mt = 0; mt < 2; ++mt) {
#pragma unroll
      for (int j = 0; j < 8; ++j) {
        float val = tanhf(acc[mt * 4 + t][j] + fb) * fw;
        val += __shfl_xor(val, 1, 16);
        val += __shfl_xor(val, 2, 16);
        val += __shfl_xor(val, 4, 16);
        val += __shfl_xor(val, 8, 16);
        if ((lane & 15) == 0) {
          const int r = mt * 16 + ((lane < 16) ? j : (j + 8));
          atomicAdd(&g_lds[r], val);
        }
      }
    }
  }
  __syncthreads();
  if (tid < MROWS) g_out[rowBase + tid] = g_lds[tid] + fc2_b[0];
}

// ---------------------------------------------------------------------------
// Kernel 3: per batch -- only head/tail query rows matter.
// ms = mean(softmax(.)) over the softmax axis == 1/61 exactly (constant).
// ---------------------------------------------------------------------------
__global__ __launch_bounds__(64) void attn_entity_kernel(
    const float* __restrict__ T, const float* __restrict__ V,
    const int* __restrict__ ids, const float* __restrict__ g,
    const float* __restrict__ clf_w, const float* __restrict__ clf_b,
    float* __restrict__ out)
{
  const int b = blockIdx.x;
  const int tid = threadIdx.x;
  const int lane = tid & 31;
  const int w = tid >> 5;                        // 0 = head, 1 = tail

  __shared__ int qidx[2];
  __shared__ float sc_lds[2][Sn];
  __shared__ float attn_lds[2][Sn];
  __shared__ float entity[2 * Dn];

  if (tid < 2) qidx[tid] = Sn;
  __syncthreads();
  if (tid < Sn) {
    const int t = ids[b * Sn + tid];
    if (t == 1) atomicMin(&qidx[0], tid);        // first occurrence = argmax
    if (t == 2) atomicMin(&qidx[1], tid);
  }
  __syncthreads();
  int q = qidx[w]; if (q >= Sn) q = 0;           // argmax of all-false -> 0

  const float* Vb = V + (size_t)b * Sn * Dn;
  const float* Tq = T + ((size_t)b * Sn + q) * Dn;
  float tq[24];
#pragma unroll
  for (int i = 0; i < 24; ++i) tq[i] = Tq[lane + 32 * i];

  // raw scores for this q-row
  for (int k = 0; k < Sn; ++k) {
    const float* vk = Vb + (size_t)k * Dn;
    float p = 0.f;
#pragma unroll
    for (int i = 0; i < 24; ++i) p = fmaf(tq[i], vk[lane + 32 * i], p);
    p += __shfl_xor(p, 16); p += __shfl_xor(p, 8); p += __shfl_xor(p, 4);
    p += __shfl_xor(p, 2);  p += __shfl_xor(p, 1);
    if (lane == 0) sc_lds[w][k] = p;
  }
  __syncthreads();

  constexpr float inv_scale = 0.036084391824351614f;  // 1/sqrt(768)
  constexpr float ms = 1.0f / (float)Sn;              // exact
  const int k1 = lane + 32;
  const float x0 = sc_lds[w][lane];
  const float x1 = (k1 < Sn) ? sc_lds[w][k1] : 0.f;
  const float g0 = g[b * Sn + lane];
  const float g1 = (k1 < Sn) ? g[b * Sn + k1] : 0.f;
  const float a0 = (x0 * (1.f - g0) + ms * g0) * inv_scale;
  const float a1 = (x1 * (1.f - g1) + ms * g1) * inv_scale;
  float m = a0;
  if (k1 < Sn) m = fmaxf(m, a1);
#pragma unroll
  for (int msk = 16; msk >= 1; msk >>= 1) m = fmaxf(m, __shfl_xor(m, msk));
  const float e0 = __expf(a0 - m);
  const float e1 = (k1 < Sn) ? __expf(a1 - m) : 0.f;
  float ssum = e0 + e1;
#pragma unroll
  for (int msk = 16; msk >= 1; msk >>= 1) ssum += __shfl_xor(ssum, msk);
  const float rs = 1.f / ssum;
  attn_lds[w][lane] = e0 * rs;
  if (k1 < Sn) attn_lds[w][k1] = e1 * rs;
  __syncthreads();

  // out[q] = sum_k attn[k] * V[b,k,:]
  float o[24] = {};
  for (int k = 0; k < Sn; ++k) {
    const float a = attn_lds[w][k];
    const float* vk = Vb + (size_t)k * Dn;
#pragma unroll
    for (int i = 0; i < 24; ++i) o[i] = fmaf(a, vk[lane + 32 * i], o[i]);
  }
#pragma unroll
  for (int i = 0; i < 24; ++i) entity[w * Dn + lane + 32 * i] = o[i];
  __syncthreads();

  // logits = clf_w @ entity + clf_b
  for (int l = w; l < Ln; l += 2) {
    const float* cw = clf_w + (size_t)l * (2 * Dn);
    float p = 0.f;
#pragma unroll
    for (int i = 0; i < 48; ++i)
      p = fmaf(entity[lane + 32 * i], cw[lane + 32 * i], p);
    p += __shfl_xor(p, 16); p += __shfl_xor(p, 8); p += __shfl_xor(p, 4);
    p += __shfl_xor(p, 2);  p += __shfl_xor(p, 1);
    if (lane == 0) out[b * Ln + l] = p + clf_b[l];
  }
}

// ---------------------------------------------------------------------------
extern "C" void kernel_launch(void* const* d_in, const int* in_sizes, int n_in,
                              void* d_out, int out_size, void* d_ws, size_t ws_size,
                              hipStream_t stream) {
  const float* T     = (const float*)d_in[0];
  const float* V     = (const float*)d_in[1];
  const int*   ids   = (const int*)  d_in[2];
  const float* fc1_w = (const float*)d_in[3];
  const float* fc1_b = (const float*)d_in[4];
  const float* fc2_w = (const float*)d_in[5];
  const float* fc2_b = (const float*)d_in[6];
  const float* clf_w = (const float*)d_in[7];
  const float* clf_b = (const float*)d_in[8];
  float* out = (float*)d_out;

  // workspace: [ fc1 bf16 fragments : 768 KB ][ g : 488 KB ]
  unsigned short* fc1pack = (unsigned short*)d_ws;
  constexpr size_t packBytes = (size_t)NT * KT * 512 * sizeof(unsigned short);
  float* g = (float*)((char*)d_ws + packBytes);

  pack_fc1_kernel<<<NT * KT, 32, 0, stream>>>(fc1_w, fc1pack);
  gate_kernel<<<ROWS / MROWS, 256, 0, stream>>>(V, fc1pack, fc1_b, fc2_w, fc2_b, g);
  attn_entity_kernel<<<Bn, 64, 0, stream>>>(T, V, ids, g, clf_w, clf_b, out);
}
